// DGCNNPartSeg_3925600109029
// MI455X (gfx1250) — matrix operand order repaired and checked by reading, not compiled, not run on hardware
//
#include <hip/hip_runtime.h>
#include <hip/hip_bf16.h>

typedef __attribute__((ext_vector_type(16))) _Float16 v16h;
typedef __attribute__((ext_vector_type(8)))  float    v8f;

#define NPTS 2048
#define KNN  20
#define NK   (NPTS * KNN)
#define EPS  1e-5f
#define SLOPE 0.2f

// ---------------------------------------------------------------------------
// Fragment layouts (wave32; this code's contiguous k order, same in A and B):
//  A frag (16x32 f16): lane = h*16+r, r = row%16; elem e -> k-local:
//      j=e/2, lo=e%2, kl = h*16+2j+lo (= h*16+e)
//  B frag (32x16 f16): lane = h*16+r, r = col%16; elem e -> k-local = h*16+e
//  Frag storage: frag_base + lane*16 + e   (512 halfs = 1 KB per fragment)
//  A frag index: (rowTile * C32 + kTile) ; B frag index: (colTile * C32 + kTile)
// ---------------------------------------------------------------------------

// squared norms: xx[n] = sum_c x[c*N+n]^2
__global__ __launch_bounds__(256) void norms_k(const float* __restrict__ x,
                                               float* __restrict__ xx, int C, int N) {
  int n = blockIdx.x * blockDim.x + threadIdx.x;
  if (n >= N) return;
  float s = 0.f;
  for (int c = 0; c < C; ++c) { float v = x[(size_t)c * N + n]; s += v * v; }
  xx[n] = s;
}

// pack W (O x C row-major f32) into A-layout f16 fragments (zero-padded)
__global__ __launch_bounds__(128) void pack_wa_k(const float* __restrict__ W,
                                                 _Float16* __restrict__ Ap,
                                                 int O, int C, int Ot16, int C32) {
  int id = blockIdx.x * blockDim.x + threadIdx.x;   // (ot*C32+kt)*32 + lane
  if (id >= Ot16 * C32 * 32) return;
  int lane = id & 31, frag = id >> 5;
  int kt = frag % C32, ot = frag / C32;
  int h = lane >> 4, r = lane & 15;
  int o = ot * 16 + r;
  _Float16* dst = Ap + (size_t)frag * 512 + lane * 16;
#pragma unroll
  for (int e = 0; e < 16; ++e) {
    int j = e >> 1, lo = e & 1;
    int kl = h * 16 + 2 * j + lo;
    int k = kt * 32 + kl;
    dst[e] = (_Float16)((o < O && k < C) ? W[(size_t)o * C + k] : 0.f);
  }
}

// pack x (C x N f32) into A-layout fragments: A[i,k] = x[k,i]
__global__ __launch_bounds__(128) void pack_xa_k(const float* __restrict__ x,
                                                 _Float16* __restrict__ Ap,
                                                 int C, int N, int C32) {
  int id = blockIdx.x * blockDim.x + threadIdx.x;
  if (id >= (N / 16) * C32 * 32) return;
  int lane = id & 31, frag = id >> 5;
  int kt = frag % C32, it = frag / C32;
  int h = lane >> 4, r = lane & 15;
  _Float16* dst = Ap + (size_t)frag * 512 + lane * 16;
#pragma unroll
  for (int e = 0; e < 16; ++e) {
    int j = e >> 1, lo = e & 1;
    int kl = h * 16 + 2 * j + lo;
    int k = kt * 32 + kl;
    dst[e] = (_Float16)((k < C) ? x[(size_t)k * N + it * 16 + r] : 0.f);
  }
}

// pack x (C x N f32) into B-layout fragments: B[k,m] = x[k,m]
__global__ __launch_bounds__(128) void pack_xb_k(const float* __restrict__ x,
                                                 _Float16* __restrict__ Bp,
                                                 int C, int N, int C32) {
  int id = blockIdx.x * blockDim.x + threadIdx.x;
  if (id >= (N / 16) * C32 * 32) return;
  int lane = id & 31, frag = id >> 5;
  int kt = frag % C32, mt = frag / C32;
  int h = lane >> 4, r = lane & 15;
  _Float16* dst = Bp + (size_t)frag * 512 + lane * 16;
#pragma unroll
  for (int e = 0; e < 16; ++e) {
    int k = kt * 32 + h * 16 + e;
    dst[e] = (_Float16)((k < C) ? x[(size_t)k * N + mt * 16 + r] : 0.f);
  }
}

// WMMA pairwise neg-dist: dist[i,j] = 2*<xi,xj> - xx[i] - xx[j]
// One wave computes a 16x64 strip (4 j-tiles); double-buffered fragment loads.
__global__ __launch_bounds__(128) void pairdist_k(const _Float16* __restrict__ Ap,
                                                  const _Float16* __restrict__ Bp,
                                                  const float* __restrict__ xx,
                                                  float* __restrict__ dist,
                                                  int C32, int N) {
  const int lane = threadIdx.x & 31;
  const int wv   = threadIdx.x >> 5;
  const int h    = lane >> 4, r = lane & 15;
  const int it   = blockIdx.y;
  const int jt0  = (blockIdx.x * 4 + wv) * 4;

  const _Float16* ap = Ap + (size_t)it * C32 * 512 + lane * 16;
  const _Float16* bp0 = Bp + (size_t)(jt0 + 0) * C32 * 512 + lane * 16;
  const _Float16* bp1 = Bp + (size_t)(jt0 + 1) * C32 * 512 + lane * 16;
  const _Float16* bp2 = Bp + (size_t)(jt0 + 2) * C32 * 512 + lane * 16;
  const _Float16* bp3 = Bp + (size_t)(jt0 + 3) * C32 * 512 + lane * 16;

  v8f acc[4] = {{}, {}, {}, {}};
  v16h a0 = *(const v16h*)ap;
  v16h b0 = *(const v16h*)bp0, b1 = *(const v16h*)bp1;
  v16h b2 = *(const v16h*)bp2, b3 = *(const v16h*)bp3;
  for (int kt = 0; kt < C32 - 1; ++kt) {
    ap += 512; bp0 += 512; bp1 += 512; bp2 += 512; bp3 += 512;
    v16h an = *(const v16h*)ap;
    v16h n0 = *(const v16h*)bp0, n1 = *(const v16h*)bp1;
    v16h n2 = *(const v16h*)bp2, n3 = *(const v16h*)bp3;
    acc[0] = __builtin_amdgcn_wmma_f32_16x16x32_f16(false, a0, false, b0, (short)0, acc[0], false, false);
    acc[1] = __builtin_amdgcn_wmma_f32_16x16x32_f16(false, a0, false, b1, (short)0, acc[1], false, false);
    acc[2] = __builtin_amdgcn_wmma_f32_16x16x32_f16(false, a0, false, b2, (short)0, acc[2], false, false);
    acc[3] = __builtin_amdgcn_wmma_f32_16x16x32_f16(false, a0, false, b3, (short)0, acc[3], false, false);
    a0 = an; b0 = n0; b1 = n1; b2 = n2; b3 = n3;
  }
  acc[0] = __builtin_amdgcn_wmma_f32_16x16x32_f16(false, a0, false, b0, (short)0, acc[0], false, false);
  acc[1] = __builtin_amdgcn_wmma_f32_16x16x32_f16(false, a0, false, b1, (short)0, acc[1], false, false);
  acc[2] = __builtin_amdgcn_wmma_f32_16x16x32_f16(false, a0, false, b2, (short)0, acc[2], false, false);
  acc[3] = __builtin_amdgcn_wmma_f32_16x16x32_f16(false, a0, false, b3, (short)0, acc[3], false, false);

#pragma unroll
  for (int t = 0; t < 4; ++t) {
#pragma unroll
    for (int rr = 0; rr < 8; ++rr) {
      int i = it * 16 + rr + 8 * h;
      int j = (jt0 + t) * 16 + r;
      dist[(size_t)i * N + j] = 2.f * acc[t][rr] - xx[i] - xx[j];
    }
  }
}

// iterative top-20 per row; one wave32 per row; row cached in registers.
__global__ __launch_bounds__(32) void topk_k(const float* __restrict__ dist,
                                             int* __restrict__ idx, int N) {
  const int lane = threadIdx.x;
  const int row  = blockIdx.x;
  const float* rp = dist + (size_t)row * N;
  float rv[64];
#pragma unroll
  for (int t = 0; t < 64; ++t) rv[t] = rp[lane + 32 * t];
  unsigned long long used = 0ull;
  for (int s = 0; s < KNN; ++s) {
    float best = -1e30f; int bt = 0;
#pragma unroll
    for (int t = 0; t < 64; ++t) {
      bool ok = !((used >> t) & 1ull);
      float v = ok ? rv[t] : -1e30f;
      if (v > best) { best = v; bt = t; }
    }
    float bv = best; int bl = lane;
    for (int off = 16; off > 0; off >>= 1) {
      float ov = __shfl_xor(bv, off, 32);
      int   ol = __shfl_xor(bl, off, 32);
      if (ov > bv || (ov == bv && ol < bl)) { bv = ov; bl = ol; }
    }
    if (lane == bl) {
      idx[(size_t)row * KNN + s] = lane + 32 * bt;
      used |= 1ull << bt;
    }
  }
}

// edge features written directly as packed B-layout f16 fragments (zero-padded)
__global__ __launch_bounds__(256) void graphfeat_k(const float* __restrict__ x,
                                                   const int* __restrict__ idx,
                                                   _Float16* __restrict__ Bp,
                                                   int C, int C32f, int N) {
  int id = blockIdx.x * blockDim.x + threadIdx.x;   // over N*K
  if (id >= NK) return;
  int n = id / KNN;
  int j = idx[id];
  int mt = id >> 4, r = id & 15;
  int Cpad = C32f * 32;
  for (int c = 0; c < Cpad; ++c) {
    float v;
    if (c < C)          v = x[(size_t)c * N + j] - x[(size_t)c * N + n];
    else if (c < 2 * C) v = x[(size_t)(c - C) * N + n];
    else                v = 0.f;
    size_t off = ((size_t)mt * C32f + (c >> 5)) * 512 +
                 (size_t)((((c >> 4) & 1) * 16 + r) * 16) + (c & 15);
    Bp[off] = (_Float16)v;
  }
}

// WMMA GEMM: Y(OxM) = act(bn(A * B)); A,B pre-packed f16 fragments.
// A row (shared by all 4 waves of the block) is staged once into LDS with the
// gfx1250 async-to-LDS path; B is double-buffered from global.
__global__ __launch_bounds__(128) void gemm_k(const _Float16* __restrict__ Ap,
                                              const _Float16* __restrict__ Bp,
                                              float* __restrict__ Yf,
                                              _Float16* __restrict__ Yh,
                                              const float* __restrict__ g,
                                              const float* __restrict__ bb,
                                              const float* __restrict__ mm,
                                              const float* __restrict__ vv,
                                              int O, int C32, int Mt, int Oc32, int fuse) {
  extern __shared__ _Float16 sA[];            // C32 fragments = C32 * 1 KB
  const int lane = threadIdx.x & 31;
  const int wv   = threadIdx.x >> 5;
  const int h    = lane >> 4, r = lane & 15;
  const int ot   = blockIdx.y;
  const int mt0  = (blockIdx.x * 4 + wv) * 4;

  // ---- async-stage A row into LDS (16B chunks, whole block cooperates)
  {
    const char* abase = (const char*)(Ap + (size_t)ot * C32 * 512);
    int chunks = C32 * 64;                    // 1 KB frag = 64 x 16B
    for (int c = threadIdx.x; c < chunks; c += 128) {
      unsigned ldsoff = (unsigned)c * 16u;
      unsigned long long ga = (unsigned long long)abase + (unsigned long long)c * 16u;
      asm volatile("global_load_async_to_lds_b128 %0, %1, off"
                   :: "v"(ldsoff), "v"(ga) : "memory");
    }
    asm volatile("s_wait_asynccnt 0x0" ::: "memory");
    __syncthreads();
  }

  const _Float16* bp0 = Bp + (size_t)(mt0 + 0) * C32 * 512 + lane * 16;
  const _Float16* bp1 = Bp + (size_t)(mt0 + 1) * C32 * 512 + lane * 16;
  const _Float16* bp2 = Bp + (size_t)(mt0 + 2) * C32 * 512 + lane * 16;
  const _Float16* bp3 = Bp + (size_t)(mt0 + 3) * C32 * 512 + lane * 16;
  const _Float16* al  = sA + lane * 16;

  v8f acc[4] = {{}, {}, {}, {}};
  v16h b0 = *(const v16h*)bp0, b1 = *(const v16h*)bp1;
  v16h b2 = *(const v16h*)bp2, b3 = *(const v16h*)bp3;
  for (int kt = 0; kt < C32 - 1; ++kt) {
    v16h a = *(const v16h*)(al); al += 512;
    bp0 += 512; bp1 += 512; bp2 += 512; bp3 += 512;
    v16h n0 = *(const v16h*)bp0, n1 = *(const v16h*)bp1;
    v16h n2 = *(const v16h*)bp2, n3 = *(const v16h*)bp3;
    acc[0] = __builtin_amdgcn_wmma_f32_16x16x32_f16(false, a, false, b0, (short)0, acc[0], false, false);
    acc[1] = __builtin_amdgcn_wmma_f32_16x16x32_f16(false, a, false, b1, (short)0, acc[1], false, false);
    acc[2] = __builtin_amdgcn_wmma_f32_16x16x32_f16(false, a, false, b2, (short)0, acc[2], false, false);
    acc[3] = __builtin_amdgcn_wmma_f32_16x16x32_f16(false, a, false, b3, (short)0, acc[3], false, false);
    b0 = n0; b1 = n1; b2 = n2; b3 = n3;
  }
  {
    v16h a = *(const v16h*)(al);
    acc[0] = __builtin_amdgcn_wmma_f32_16x16x32_f16(false, a, false, b0, (short)0, acc[0], false, false);
    acc[1] = __builtin_amdgcn_wmma_f32_16x16x32_f16(false, a, false, b1, (short)0, acc[1], false, false);
    acc[2] = __builtin_amdgcn_wmma_f32_16x16x32_f16(false, a, false, b2, (short)0, acc[2], false, false);
    acc[3] = __builtin_amdgcn_wmma_f32_16x16x32_f16(false, a, false, b3, (short)0, acc[3], false, false);
  }

  const int M = Mt * 16;
#pragma unroll
  for (int t = 0; t < 4; ++t) {
    int mt = mt0 + t;
    int m  = mt * 16 + r;
#pragma unroll
    for (int rr = 0; rr < 8; ++rr) {
      int o = ot * 16 + rr + 8 * h;
      if (o >= O) continue;
      float y = acc[t][rr];
      if (fuse) {
        float inv = g[o] * rsqrtf(vv[o] + EPS);
        y = (y - mm[o]) * inv + bb[o];
        y = (y > 0.f) ? y : SLOPE * y;
      }
      if (Yf) Yf[(size_t)o * M + m] = y;
      if (Yh) {
        size_t off = ((size_t)mt * Oc32 + (o >> 5)) * 512 +
                     (size_t)((((o >> 4) & 1) * 16 + r) * 16) + (o & 15);
        Yh[off] = (_Float16)y;
      }
    }
  }
}

// max over K neighbors: out[c,n] = max_k in[c, n*K+k]
__global__ __launch_bounds__(256) void maxk_k(const float* __restrict__ in,
                                              float* __restrict__ out, int C, int N) {
  int id = blockIdx.x * blockDim.x + threadIdx.x;
  if (id >= C * N) return;
  int c = id / N, n = id % N;
  const float* p = in + (size_t)c * NK + (size_t)n * KNN;
  float m = p[0];
#pragma unroll
  for (int k = 1; k < KNN; ++k) m = fmaxf(m, p[k]);
  out[(size_t)c * N + n] = m;
}

// cat = [x1;x2;x3] (192 x N) written as packed B-layout f16
__global__ __launch_bounds__(256) void packcat_k(const float* __restrict__ x1,
                                                 const float* __restrict__ x2,
                                                 const float* __restrict__ x3,
                                                 _Float16* __restrict__ Bp, int N) {
  int id = blockIdx.x * blockDim.x + threadIdx.x;
  if (id >= 192 * N) return;
  int c = id / N, m = id % N;
  float v = (c < 64) ? x1[(size_t)c * N + m]
          : (c < 128) ? x2[(size_t)(c - 64) * N + m]
                      : x3[(size_t)(c - 128) * N + m];
  size_t off = ((size_t)(m >> 4) * 6 + (c >> 5)) * 512 +
               (size_t)((((c >> 4) & 1) * 16 + (m & 15)) * 16) + (c & 15);
  Bp[off] = (_Float16)v;
}

// gmax[c] = max_n G[c,n]
__global__ __launch_bounds__(256) void gmax_k(const float* __restrict__ G,
                                              float* __restrict__ gmax, int N) {
  int c = blockIdx.x * blockDim.x + threadIdx.x;
  if (c >= 1024) return;
  const float* p = G + (size_t)c * N;
  float m = p[0];
  for (int n = 1; n < N; ++n) m = fmaxf(m, p[n]);
  gmax[c] = m;
}

// h = [bcast(gmax); x1; x2; x3] (1216 x N) written as packed B-layout f16
__global__ __launch_bounds__(256) void packh_k(const float* __restrict__ gmax,
                                               const float* __restrict__ x1,
                                               const float* __restrict__ x2,
                                               const float* __restrict__ x3,
                                               _Float16* __restrict__ Bp, int N) {
  int id = blockIdx.x * blockDim.x + threadIdx.x;
  if (id >= 1216 * N) return;
  int c = id / N, m = id % N;
  float v;
  if (c < 1024)      v = gmax[c];
  else if (c < 1088) v = x1[(size_t)(c - 1024) * N + m];
  else if (c < 1152) v = x2[(size_t)(c - 1088) * N + m];
  else               v = x3[(size_t)(c - 1152) * N + m];
  size_t off = ((size_t)(m >> 4) * 38 + (c >> 5)) * 512 +
               (size_t)((((c >> 4) & 1) * 16 + (m & 15)) * 16) + (c & 15);
  Bp[off] = (_Float16)v;
}

// ---------------------------------------------------------------------------
extern "C" void kernel_launch(void* const* d_in, const int* in_sizes, int n_in,
                              void* d_out, int out_size, void* d_ws, size_t ws_size,
                              hipStream_t stream) {
  (void)in_sizes; (void)n_in; (void)out_size; (void)ws_size;
  const int N = NPTS, B = 8;

  const float* x = (const float*)d_in[0];
  struct BN { const float *g, *b, *m, *v; };
  auto P   = [&](int i) { return (const float*)d_in[i]; };
  auto BNP = [&](int i) { BN p; p.g = P(i); p.b = P(i+1); p.m = P(i+2); p.v = P(i+3); return p; };
  const float *W1 = P(1);  BN bn1 = BNP(2);
  const float *W2 = P(6);  BN bn2 = BNP(7);
  const float *W3 = P(11); BN bn3 = BNP(12);
  const float *W4 = P(16); BN bn4 = BNP(17);
  const float *W5 = P(21); BN bn5 = BNP(22);
  const float *W6 = P(26); BN bn6 = BNP(27);
  const float *W8 = P(31); BN bn8 = BNP(32);
  const float *W9 = P(36); BN bn9 = BNP(37);
  const float *W10 = P(41); BN bn10 = BNP(42);
  const float *W11 = P(46);
  float* out = (float*)d_out;

  // ---- workspace carving: f32 region then f16 region
  float* fp = (float*)d_ws;
  auto falloc = [&](size_t n) { float* p = fp; fp += n; return p; };
  float* DIST = falloc((size_t)N * N);
  float* T2   = falloc((size_t)64 * NK);
  float* X1   = falloc((size_t)B * 64 * N);
  float* X2   = falloc((size_t)B * 64 * N);
  float* X3   = falloc((size_t)B * 64 * N);
  float* G    = falloc((size_t)1024 * N);
  float* GMX  = falloc(1024);
  float* XX   = falloc(N);
  int*   IDX  = (int*)falloc(NK);

  _Float16* hp = (_Float16*)fp;
  auto halloc = [&](size_t n) { _Float16* p = hp; hp += n; return p; };
  _Float16* W1p  = halloc(4 * 1 * 512);
  _Float16* W2p  = halloc(4 * 2 * 512);
  _Float16* W3p  = halloc(4 * 4 * 512);
  _Float16* W4p  = halloc(4 * 2 * 512);
  _Float16* W5p  = halloc(4 * 4 * 512);
  _Float16* W6p  = halloc(64 * 6 * 512);
  _Float16* W8p  = halloc(16 * 38 * 512);
  _Float16* W9p  = halloc(16 * 8 * 512);
  _Float16* W10p = halloc(8 * 8 * 512);
  _Float16* W11p = halloc(4 * 4 * 512);
  _Float16* FEATp = halloc((size_t)2560 * 4 * 512);  // up to 128 ch, Mt=2560
  _Float16* T1p   = halloc((size_t)2560 * 2 * 512);  // 64 ch
  _Float16* CATp  = halloc((size_t)128 * 6 * 512);   // 192 ch, Mt=128
  _Float16* Hp    = halloc((size_t)128 * 38 * 512);  // 1216 ch
  _Float16* H1p   = halloc((size_t)128 * 8 * 512);   // 256 ch
  _Float16* H2p   = halloc((size_t)128 * 8 * 512);
  _Float16* H3p   = halloc((size_t)128 * 4 * 512);   // 128 ch
  _Float16* XAp   = halloc((size_t)128 * 2 * 512);   // x as A-frags (<=64 ch)
  _Float16* XBp   = halloc((size_t)128 * 2 * 512);   // x as B-frags

  auto gemm = [&](const _Float16* Ap, const _Float16* Bp, float* Yf, _Float16* Yh,
                  BN bn, int O, int C32, int Mt, int fuse) {
    dim3 grid(Mt / 16, (O + 15) / 16);
    size_t shmem = (size_t)C32 * 1024;   // A row staged in LDS
    gemm_k<<<grid, 128, shmem, stream>>>(Ap, Bp, Yf, Yh, bn.g, bn.b, bn.m, bn.v,
                                         O, C32, Mt, O >> 5, fuse);
  };
  auto packw = [&](const float* Wp, _Float16* Ap, int O, int C, int C32) {
    int total = ((O + 15) / 16) * C32 * 32;
    pack_wa_k<<<(total + 127) / 128, 128, 0, stream>>>(Wp, Ap, O, C, (O + 15) / 16, C32);
  };
  auto knn_stage = [&](const float* xb, int C, int C32) {
    norms_k<<<N / 256, 256, 0, stream>>>(xb, XX, C, N);
    int total = (N / 16) * C32 * 32;
    pack_xa_k<<<(total + 127) / 128, 128, 0, stream>>>(xb, XAp, C, N, C32);
    pack_xb_k<<<(total + 127) / 128, 128, 0, stream>>>(xb, XBp, C, N, C32);
    pairdist_k<<<dim3(N / 16 / 16, N / 16), 128, 0, stream>>>(XAp, XBp, XX, DIST, C32, N);
    topk_k<<<N, 32, 0, stream>>>(DIST, IDX, N);
  };
  BN nobn = {bn1.g, bn1.b, bn1.m, bn1.v};  // unused when fuse==0

  // pack all weights once (stream-ordered before first use)
  packw(W1, W1p, 64, 6, 1);     packw(W2, W2p, 64, 64, 2);
  packw(W3, W3p, 64, 128, 4);   packw(W4, W4p, 64, 64, 2);
  packw(W5, W5p, 64, 128, 4);   packw(W6, W6p, 1024, 192, 6);
  packw(W8, W8p, 256, 1216, 38); packw(W9, W9p, 256, 256, 8);
  packw(W10, W10p, 128, 256, 8); packw(W11, W11p, 50, 128, 4);

  for (int b = 0; b < B; ++b) {
    const float* xb = x + (size_t)b * 3 * N;
    float* x1b = X1 + (size_t)b * 64 * N;
    float* x2b = X2 + (size_t)b * 64 * N;
    float* x3b = X3 + (size_t)b * 64 * N;

    // ---- edge-conv 1 (C=3 -> 6 -> 64 -> 64, max over K)
    knn_stage(xb, 3, 1);
    graphfeat_k<<<NK / 256, 256, 0, stream>>>(xb, IDX, FEATp, 3, 1, N);
    gemm(W1p, FEATp, nullptr, T1p, bn1, 64, 1, 2560, 1);
    gemm(W2p, T1p, T2, nullptr, bn2, 64, 2, 2560, 1);
    maxk_k<<<(64 * N) / 256, 256, 0, stream>>>(T2, x1b, 64, N);

    // ---- edge-conv 2
    knn_stage(x1b, 64, 2);
    graphfeat_k<<<NK / 256, 256, 0, stream>>>(x1b, IDX, FEATp, 64, 4, N);
    gemm(W3p, FEATp, nullptr, T1p, bn3, 64, 4, 2560, 1);
    gemm(W4p, T1p, T2, nullptr, bn4, 64, 2, 2560, 1);
    maxk_k<<<(64 * N) / 256, 256, 0, stream>>>(T2, x2b, 64, N);

    // ---- edge-conv 3
    knn_stage(x2b, 64, 2);
    graphfeat_k<<<NK / 256, 256, 0, stream>>>(x2b, IDX, FEATp, 64, 4, N);
    gemm(W5p, FEATp, T2, nullptr, bn5, 64, 4, 2560, 1);
    maxk_k<<<(64 * N) / 256, 256, 0, stream>>>(T2, x3b, 64, N);

    // ---- global feature + segmentation head
    packcat_k<<<(192 * N) / 256, 256, 0, stream>>>(x1b, x2b, x3b, CATp, N);
    gemm(W6p, CATp, G, nullptr, bn6, 1024, 6, 128, 1);
    gmax_k<<<4, 256, 0, stream>>>(G, GMX, N);
    packh_k<<<(1216 * N) / 256, 256, 0, stream>>>(GMX, x1b, x2b, x3b, Hp, N);
    gemm(W8p, Hp, nullptr, H1p, bn8, 256, 38, 128, 1);
    gemm(W9p, H1p, nullptr, H2p, bn9, 256, 8, 128, 1);
    gemm(W10p, H2p, nullptr, H3p, bn10, 128, 8, 128, 1);
    gemm(W11p, H3p, out + (size_t)b * 50 * N, nullptr, nobn, 50, 4, 128, 0);
  }
}
